// DeformableTransformerEncoderLayer_55319178772977
// MI455X (gfx1250) — compile-verified, matrix operand order checked by reading, not run on hardware
//
#include <hip/hip_runtime.h>
#include <hip/hip_bf16.h>

// ---------------- problem constants (static per reference) ----------------
#define BQ      8
#define LEN_T   5440            // 64*64 + 32*32 + 16*16 + 8*8
#define MROWS   (BQ * LEN_T)    // 43520 token rows (multiple of 32)
#define DMODEL  256
#define DFFN    1024
#define NHEADS  8
#define NLEV    4
#define NPTS    4
#define HDIM    32

typedef unsigned short ushortT;
typedef __attribute__((ext_vector_type(16))) __bf16 v16bf;
typedef __attribute__((ext_vector_type(8)))  float  v8f;

union Frag { v16bf v; uint4 q[2]; };

static __device__ __forceinline__ ushortT f2bf(float f) {
    union { float f; unsigned u; } v; v.f = f;
    unsigned r = v.u + 0x7FFFu + ((v.u >> 16) & 1u);   // round-to-nearest-even
    return (ushortT)(r >> 16);
}

// ---------------- pack: transpose fp32 W[K,N] -> bf16 Wt[N,K] ----------------
__global__ void pack_wt_kernel(const float* __restrict__ W, ushortT* __restrict__ Wt,
                               int K, int N) {
    int i = blockIdx.x * blockDim.x + threadIdx.x;
    if (i >= K * N) return;
    int n = i / K, k = i - n * K;
    Wt[i] = f2bf(W[(size_t)k * N + n]);
}

// ---------------- pack: q = bf16(src+pos), s = bf16(src) ----------------
__global__ void add_pack_kernel(const float* __restrict__ src, const float* __restrict__ pos,
                                ushortT* __restrict__ qbf, ushortT* __restrict__ sbf, int n) {
    int i = blockIdx.x * blockDim.x + threadIdx.x;
    if (i >= n) return;
    float s = src[i];
    qbf[i] = f2bf(s + pos[i]);
    sbf[i] = f2bf(s);
}

// ---------------- WMMA bf16 GEMM: C[M,N] = A[M,K] * Bt[N,K]^T + bias ----------------
// block = 128 threads (4 waves); each wave computes a 32x32 tile (2x2 WMMA tiles,
// register-blocked: 4 fragment loads -> 4 WMMAs per k-step).
// grid = (M/32, N/128)
static __device__ __forceinline__ void store_tile(const v8f& acc, int tm, int tn, int lane,
                                                  const float* __restrict__ bias,
                                                  float* __restrict__ Cf,
                                                  ushortT* __restrict__ Cbf,
                                                  int N, int relu) {
    int r    = lane & 15;
    int col  = tn + r;
    float bv = bias ? bias[col] : 0.0f;
    int row0 = tm + ((lane >> 4) << 3);
    #pragma unroll
    for (int i = 0; i < 8; ++i) {
        float v = acc[i] + bv;
        if (relu) v = fmaxf(v, 0.0f);
        size_t idx = (size_t)(row0 + i) * N + col;
        if (Cf)  Cf[idx]  = v;
        if (Cbf) Cbf[idx] = f2bf(v);
    }
}

__global__ void gemm_wmma_bf16(const ushortT* __restrict__ A, const ushortT* __restrict__ Bt,
                               const float* __restrict__ bias,
                               float* __restrict__ Cf, ushortT* __restrict__ Cbf,
                               int M, int N, int K, int relu) {
    int lane = threadIdx.x & 31;
    int w    = threadIdx.x >> 5;
    int tm   = blockIdx.x * 32;
    int tn   = (blockIdx.y * 4 + w) * 32;
    int r    = lane & 15;
    int kb   = (lane >> 4) << 3;            // 0 or 8: ISA 16-bit fragment K base

    const ushortT* pa0 = A  + (size_t)(tm + r) * K + kb;
    const ushortT* pa1 = pa0 + (size_t)16 * K;
    const ushortT* pb0 = Bt + (size_t)(tn + r) * K + kb;
    const ushortT* pb1 = pb0 + (size_t)16 * K;

    v8f acc00 = {}, acc01 = {}, acc10 = {}, acc11 = {};
    for (int k = 0; k < K; k += 32) {
        Frag a0, a1, b0, b1;
        a0.q[0] = *(const uint4*)(pa0);  a0.q[1] = *(const uint4*)(pa0 + 16);
        a1.q[0] = *(const uint4*)(pa1);  a1.q[1] = *(const uint4*)(pa1 + 16);
        b0.q[0] = *(const uint4*)(pb0);  b0.q[1] = *(const uint4*)(pb0 + 16);
        b1.q[0] = *(const uint4*)(pb1);  b1.q[1] = *(const uint4*)(pb1 + 16);
        __builtin_prefetch(pa0 + 64, 0, 1);
        __builtin_prefetch(pb0 + 64, 0, 1);
        pa0 += 32; pa1 += 32; pb0 += 32; pb1 += 32;
        acc00 = __builtin_amdgcn_wmma_f32_16x16x32_bf16(false, a0.v, false, b0.v, (short)0, acc00, false, false);
        acc01 = __builtin_amdgcn_wmma_f32_16x16x32_bf16(false, a0.v, false, b1.v, (short)0, acc01, false, false);
        acc10 = __builtin_amdgcn_wmma_f32_16x16x32_bf16(false, a1.v, false, b0.v, (short)0, acc10, false, false);
        acc11 = __builtin_amdgcn_wmma_f32_16x16x32_bf16(false, a1.v, false, b1.v, (short)0, acc11, false, false);
    }

    store_tile(acc00, tm,      tn,      lane, bias, Cf, Cbf, N, relu);
    store_tile(acc01, tm,      tn + 16, lane, bias, Cf, Cbf, N, relu);
    store_tile(acc10, tm + 16, tn,      lane, bias, Cf, Cbf, N, relu);
    store_tile(acc11, tm + 16, tn + 16, lane, bias, Cf, Cbf, N, relu);
}

// ---------------- per-(row,head) softmax over 16 logits, in place ----------------
__global__ void softmax16_kernel(float* __restrict__ a, int nrh) {
    int t = blockIdx.x * blockDim.x + threadIdx.x;
    if (t >= nrh) return;
    float* p = a + (size_t)t * 16;
    float m = p[0];
    #pragma unroll
    for (int i = 1; i < 16; ++i) m = fmaxf(m, p[i]);
    float s = 0.f, e[16];
    #pragma unroll
    for (int i = 0; i < 16; ++i) { e[i] = __expf(p[i] - m); s += e[i]; }
    float inv = 1.0f / s;
    #pragma unroll
    for (int i = 0; i < 16; ++i) p[i] = e[i] * inv;
}

// ---------------- deformable sampling: one wave per (row, head), lane = channel ----
__global__ void ms_sample_kernel(const float* __restrict__ value,
                                 const float* __restrict__ off,
                                 const float* __restrict__ attn,
                                 const float* __restrict__ ref,
                                 ushortT* __restrict__ out_bf) {
    int gid = blockIdx.x * blockDim.x + threadIdx.x;
    int wid = gid >> 5;
    int d   = gid & 31;
    if (wid >= MROWS * NHEADS) return;
    int row = wid >> 3;
    int h   = wid & 7;
    int b   = row / LEN_T;

    const int sz[NLEV]  = {64, 32, 16, 8};
    const int lst[NLEV] = {0, 4096, 5120, 5376};

    const float* refp  = ref  + (size_t)row * NLEV * 2;
    const float* offp  = off  + (size_t)row * DMODEL + h * (NLEV * NPTS * 2);
    const float* attp  = attn + (size_t)row * (NHEADS * NLEV * NPTS) + h * (NLEV * NPTS);
    const float* vbase = value + (size_t)b * LEN_T * DMODEL + h * HDIM + d;

    float acc = 0.0f;
    #pragma unroll
    for (int l = 0; l < NLEV; ++l) {
        int   Wi = sz[l], Hi = sz[l];
        float Wf = (float)Wi, Hf = (float)Hi;
        float rx = refp[l * 2 + 0], ry = refp[l * 2 + 1];
        const float* vlev = vbase + (size_t)lst[l] * DMODEL;
        #pragma unroll
        for (int p = 0; p < NPTS; ++p) {
            float ox = offp[(l * NPTS + p) * 2 + 0];
            float oy = offp[(l * NPTS + p) * 2 + 1];
            float x  = (rx + ox / Wf) * Wf - 0.5f;
            float y  = (ry + oy / Hf) * Hf - 0.5f;
            float x0 = floorf(x), y0 = floorf(y);
            float wx = x - x0,    wy = y - y0;
            int x0i = (int)x0, y0i = (int)y0;
            float s = 0.0f;
            #pragma unroll
            for (int cy = 0; cy < 2; ++cy) {
                int yi = y0i + cy;
                float wyv = cy ? wy : (1.0f - wy);
                #pragma unroll
                for (int cx = 0; cx < 2; ++cx) {
                    int xi = x0i + cx;
                    if (xi >= 0 && xi < Wi && yi >= 0 && yi < Hi) {
                        float wv = (cx ? wx : (1.0f - wx)) * wyv;
                        s += wv * vlev[(size_t)(yi * Wi + xi) * DMODEL];
                    }
                }
            }
            acc += attp[l * NPTS + p] * s;
        }
    }
    out_bf[(size_t)row * DMODEL + h * HDIM + d] = f2bf(acc);
}

// ---------------- residual + layernorm: one wave per row (256 cols) ----------------
__global__ void resid_ln_kernel(const float* __restrict__ x, const float* __restrict__ y,
                                const float* __restrict__ g, const float* __restrict__ beta,
                                float* __restrict__ outf, ushortT* __restrict__ outbf,
                                int nrows) {
    int row  = blockIdx.x * (blockDim.x >> 5) + (threadIdx.x >> 5);
    int lane = threadIdx.x & 31;
    if (row >= nrows) return;
    const float* px = x + (size_t)row * DMODEL;
    const float* py = y + (size_t)row * DMODEL;
    float v[8];
    float s = 0.0f;
    #pragma unroll
    for (int i = 0; i < 8; ++i) { v[i] = px[lane + i * 32] + py[lane + i * 32]; s += v[i]; }
    #pragma unroll
    for (int o = 16; o > 0; o >>= 1) s += __shfl_xor(s, o, 32);
    float mean = s * (1.0f / DMODEL);
    float var = 0.0f;
    #pragma unroll
    for (int i = 0; i < 8; ++i) { float t = v[i] - mean; var += t * t; }
    #pragma unroll
    for (int o = 16; o > 0; o >>= 1) var += __shfl_xor(var, o, 32);
    var *= (1.0f / DMODEL);
    float inv = rsqrtf(var + 1e-5f);
    #pragma unroll
    for (int i = 0; i < 8; ++i) {
        int c = lane + i * 32;
        float o = (v[i] - mean) * inv * g[c] + beta[c];
        if (outf)  outf[(size_t)row * DMODEL + c]  = o;
        if (outbf) outbf[(size_t)row * DMODEL + c] = f2bf(o);
    }
}

// ---------------- passthrough queries into output tail ----------------
__global__ void tail_copy_kernel(const float* __restrict__ sq, const float* __restrict__ cq,
                                 float* __restrict__ out) {
    int i = blockIdx.x * blockDim.x + threadIdx.x;
    int nq = BQ * DMODEL;                 // 2048
    if (i < nq)            out[i]      = sq[i];
    else if (i < 2 * nq)   out[i]      = cq[i - nq];
}

// ======================================================================
extern "C" void kernel_launch(void* const* d_in, const int* in_sizes, int n_in,
                              void* d_out, int out_size, void* d_ws, size_t ws_size,
                              hipStream_t stream) {
    (void)in_sizes; (void)n_in; (void)out_size; (void)ws_size;

    const float* src = (const float*)d_in[0];
    const float* sq  = (const float*)d_in[1];
    const float* cq  = (const float*)d_in[2];
    const float* pos = (const float*)d_in[3];
    const float* ref = (const float*)d_in[4];
    // d_in[5]=spatial_shapes, d_in[6]=level_start_index : static, hardcoded
    const float* vw = (const float*)d_in[7];   const float* vb = (const float*)d_in[8];
    const float* ow = (const float*)d_in[9];   const float* ob = (const float*)d_in[10];
    const float* aw = (const float*)d_in[11];  const float* ab = (const float*)d_in[12];
    const float* pw = (const float*)d_in[13];  const float* pb = (const float*)d_in[14];
    const float* n1g = (const float*)d_in[15]; const float* n1b = (const float*)d_in[16];
    const float* l1w = (const float*)d_in[17]; const float* l1b = (const float*)d_in[18];
    const float* l2w = (const float*)d_in[19]; const float* l2b = (const float*)d_in[20];
    const float* n2g = (const float*)d_in[21]; const float* n2b = (const float*)d_in[22];
    float* out = (float*)d_out;

    // ---- workspace carve-out (512B aligned) ----
    char* base = (char*)d_ws;
    size_t cur = 0;
    auto alloc = [&](size_t bytes) -> char* {
        char* p = base + cur;
        cur += (bytes + 511) & ~(size_t)511;
        return p;
    };
    const size_t MD  = (size_t)MROWS * DMODEL;
    ushortT* qbf   = (ushortT*)alloc(MD * 2);
    ushortT* sbf   = (ushortT*)alloc(MD * 2);
    ushortT* wVt   = (ushortT*)alloc((size_t)DMODEL * DMODEL * 2);
    ushortT* wOt   = (ushortT*)alloc((size_t)DMODEL * DMODEL * 2);
    ushortT* wAt   = (ushortT*)alloc((size_t)128 * DMODEL * 2);
    ushortT* wPt   = (ushortT*)alloc((size_t)DMODEL * DMODEL * 2);
    ushortT* wL1t  = (ushortT*)alloc((size_t)DFFN * DMODEL * 2);
    ushortT* wL2t  = (ushortT*)alloc((size_t)DMODEL * DFFN * 2);
    float*   value = (float*)alloc(MD * 4);
    float*   offb  = (float*)alloc(MD * 4);
    float*   attnb = (float*)alloc((size_t)MROWS * 128 * 4);
    ushortT* oabf  = (ushortT*)alloc(MD * 2);
    float*   pwout = (float*)alloc(MD * 4);
    float*   x1f   = (float*)alloc(MD * 4);
    ushortT* x1bf  = (ushortT*)alloc(MD * 2);
    ushortT* hbf   = (ushortT*)alloc((size_t)MROWS * DFFN * 2);
    float*   ffnf  = (float*)alloc(MD * 4);

    // ---- weight packing (transpose + bf16) ----
    auto packw = [&](const float* W, ushortT* Wt, int K, int N) {
        int n = K * N;
        hipLaunchKernelGGL(pack_wt_kernel, dim3((n + 255) / 256), dim3(256), 0, stream, W, Wt, K, N);
    };
    packw(vw,  wVt,  DMODEL, DMODEL);
    packw(ow,  wOt,  DMODEL, DMODEL);
    packw(aw,  wAt,  DMODEL, 128);
    packw(pw,  wPt,  DMODEL, DMODEL);
    packw(l1w, wL1t, DMODEL, DFFN);
    packw(l2w, wL2t, DFFN,   DMODEL);

    // ---- activation packing ----
    {
        int n = (int)MD;
        hipLaunchKernelGGL(add_pack_kernel, dim3((n + 255) / 256), dim3(256), 0, stream,
                           src, pos, qbf, sbf, n);
    }

    // ---- GEMMs: value / sampling offsets / attention logits ----
    dim3 gblk(128);
    hipLaunchKernelGGL(gemm_wmma_bf16, dim3(MROWS / 32, DMODEL / 128), gblk, 0, stream,
                       sbf, wVt, vb, value, (ushortT*)nullptr, MROWS, DMODEL, DMODEL, 0);
    hipLaunchKernelGGL(gemm_wmma_bf16, dim3(MROWS / 32, DMODEL / 128), gblk, 0, stream,
                       qbf, wOt, ob, offb, (ushortT*)nullptr, MROWS, DMODEL, DMODEL, 0);
    hipLaunchKernelGGL(gemm_wmma_bf16, dim3(MROWS / 32, 128 / 128), gblk, 0, stream,
                       qbf, wAt, ab, attnb, (ushortT*)nullptr, MROWS, 128, DMODEL, 0);

    // ---- softmax over the 16 (level,point) logits per head ----
    {
        int nrh = MROWS * NHEADS;
        hipLaunchKernelGGL(softmax16_kernel, dim3((nrh + 255) / 256), dim3(256), 0, stream,
                           attnb, nrh);
    }

    // ---- bilinear deformable sampling (writes bf16 for pw GEMM) ----
    {
        long long threads = (long long)MROWS * NHEADS * 32;
        hipLaunchKernelGGL(ms_sample_kernel, dim3((unsigned)((threads + 255) / 256)), dim3(256),
                           0, stream, value, offb, attnb, ref, oabf);
    }

    // ---- output projection ----
    hipLaunchKernelGGL(gemm_wmma_bf16, dim3(MROWS / 32, DMODEL / 128), gblk, 0, stream,
                       oabf, wPt, pb, pwout, (ushortT*)nullptr, MROWS, DMODEL, DMODEL, 0);

    // ---- residual + LN #1 ----
    hipLaunchKernelGGL(resid_ln_kernel, dim3(MROWS / 8), dim3(256), 0, stream,
                       src, pwout, n1g, n1b, x1f, x1bf, MROWS);

    // ---- FFN ----
    hipLaunchKernelGGL(gemm_wmma_bf16, dim3(MROWS / 32, DFFN / 128), gblk, 0, stream,
                       x1bf, wL1t, l1b, (float*)nullptr, hbf, MROWS, DFFN, DMODEL, 1);
    hipLaunchKernelGGL(gemm_wmma_bf16, dim3(MROWS / 32, DMODEL / 128), gblk, 0, stream,
                       hbf, wL2t, l2b, ffnf, (ushortT*)nullptr, MROWS, DMODEL, DFFN, 0);

    // ---- residual + LN #2 -> final output ----
    hipLaunchKernelGGL(resid_ln_kernel, dim3(MROWS / 8), dim3(256), 0, stream,
                       x1f, ffnf, n2g, n2b, out, (ushortT*)nullptr, MROWS);

    // ---- passthrough queries ----
    hipLaunchKernelGGL(tail_copy_kernel, dim3((2 * BQ * DMODEL + 255) / 256), dim3(256), 0, stream,
                       sq, cq, out + MD);
}
